// Fp8Linear_7876970021128
// MI455X (gfx1250) — 8-bit matrix operand order repaired and checked by reading, compiled, not run on hardware
//
#include <hip/hip_runtime.h>


typedef __attribute__((ext_vector_type(16))) int   v16i;
typedef __attribute__((ext_vector_type(8)))  float v8f;
typedef int v4i_vs __attribute__((vector_size(16)));

union AFrag { v16i v; uint2 h[8]; };
union BFrag { v16i v; uint2 h[8]; };

#if __has_builtin(__builtin_amdgcn_global_load_async_to_lds_b128) && \
    __has_builtin(__builtin_amdgcn_s_wait_asynccnt)
#define USE_ASYNC_LDS 1
#else
#define USE_ASYNC_LDS 0
#endif

#define GLOBAL_AS __attribute__((address_space(1)))
#define LDS_AS    __attribute__((address_space(3)))

__global__ void fp8_quantize_kernel(const float* __restrict__ x,
                                    const float* __restrict__ w,
                                    const float* __restrict__ scale_ptr,
                                    unsigned int* __restrict__ qx,
                                    unsigned int* __restrict__ qw,
                                    long long nx4, long long nw4) {
    const float s = *scale_ptr;
    const long long tid    = (long long)blockIdx.x * blockDim.x + threadIdx.x;
    const long long stride = (long long)gridDim.x * blockDim.x;

    const float4* __restrict__ x4 = (const float4*)x;
    const float4* __restrict__ w4 = (const float4*)w;

    for (long long i = tid; i < nx4; i += stride) {
        float4 v = x4[i];
        float a = fminf(fmaxf(v.x * s, -0.5f), 0.5f);
        float b = fminf(fmaxf(v.y * s, -0.5f), 0.5f);
        float c = fminf(fmaxf(v.z * s, -0.5f), 0.5f);
        float d = fminf(fmaxf(v.w * s, -0.5f), 0.5f);
        int lo = __builtin_amdgcn_cvt_pk_fp8_f32(a, b, 0, false);
        int hi = __builtin_amdgcn_cvt_pk_fp8_f32(c, d, lo, true);
        qx[i] = (unsigned int)hi;
    }
    for (long long i = tid; i < nw4; i += stride) {
        float4 v = w4[i];
        int lo = __builtin_amdgcn_cvt_pk_fp8_f32(v.x, v.y, 0, false);
        int hi = __builtin_amdgcn_cvt_pk_fp8_f32(v.z, v.w, lo, true);
        qw[i] = (unsigned int)hi;
    }
}

#define BM 128
#define BN 128
#define BK 128
#if USE_ASYNC_LDS
#define NBUF 2
#else
#define NBUF 1
#endif

__global__ __launch_bounds__(256)
void fp8_gemm_wmma_kernel(const unsigned char* __restrict__ qx,
                          const unsigned char* __restrict__ qw,
                          const float* __restrict__ bias,
                          float* __restrict__ out,
                          int M, int N, int K) {
    __shared__ __align__(16) unsigned char As[NBUF][BM * BK];
    __shared__ __align__(16) unsigned char Bs[NBUF][BN * BK];

    const int tid  = threadIdx.x;
    const int lane = tid & 31;
    const int wave = tid >> 5;

    const int m0 = blockIdx.y * BM;
    const int n0 = blockIdx.x * BN;

    const int wm = (wave >> 2) * 64;
    const int wn = (wave & 3) * 32;

    v8f acc[4][2] = {};

    const int lrow = tid >> 3;
    const int lcol = (tid & 7) * 16;

    const int frow  = lane & 15;
    const int abase = (lane >> 4) * 8;
    const int bbase = (lane >> 4) * 8;

#if USE_ASYNC_LDS
    auto fill_async = [&](int buf, int kk) {
        #pragma unroll
        for (int p = 0; p < 4; ++p) {
            const int r = lrow + p * 32;
            __builtin_amdgcn_global_load_async_to_lds_b128(
                (GLOBAL_AS v4i_vs*)(qx + (size_t)(m0 + r) * K + kk + lcol),
                (LDS_AS v4i_vs*)(&As[buf][r * BK + lcol]), 0, 0);
        }
        #pragma unroll
        for (int p = 0; p < 4; ++p) {
            const int r = lrow + p * 32;
            __builtin_amdgcn_global_load_async_to_lds_b128(
                (GLOBAL_AS v4i_vs*)(qw + (size_t)(n0 + r) * K + kk + lcol),
                (LDS_AS v4i_vs*)(&Bs[buf][r * BK + lcol]), 0, 0);
        }
    };
#else
    auto fill_sync = [&](int buf, int kk) {
        #pragma unroll
        for (int p = 0; p < 4; ++p) {
            const int r = lrow + p * 32;
            uint4 d = *(const uint4*)(qx + (size_t)(m0 + r) * K + kk + lcol);
            *(uint4*)(&As[buf][r * BK + lcol]) = d;
        }
        #pragma unroll
        for (int p = 0; p < 4; ++p) {
            const int r = lrow + p * 32;
            uint4 d = *(const uint4*)(qw + (size_t)(n0 + r) * K + kk + lcol);
            *(uint4*)(&Bs[buf][r * BK + lcol]) = d;
        }
    };
#endif

    auto compute = [&](int buf) {
        AFrag a[4];
        #pragma unroll
        for (int mt = 0; mt < 4; ++mt) {
            const unsigned char* row = &As[buf][(wm + mt * 16 + frow) * BK + abase];
            #pragma unroll
            for (int c = 0; c < 8; ++c)
                a[mt].h[c] = *(const uint2*)(row + c * 16);
        }
        BFrag b[2];
        #pragma unroll
        for (int nt = 0; nt < 2; ++nt) {
            const unsigned char* row = &Bs[buf][(wn + nt * 16 + frow) * BK + bbase];
            #pragma unroll
            for (int c = 0; c < 8; ++c)
                b[nt].h[c] = *(const uint2*)(row + c * 16);
        }
        #pragma unroll
        for (int mt = 0; mt < 4; ++mt)
            #pragma unroll
            for (int nt = 0; nt < 2; ++nt)
                acc[mt][nt] = __builtin_amdgcn_wmma_f32_16x16x128_fp8_fp8(
                    a[mt].v, b[nt].v, (short)0, acc[mt][nt],
                     false,  false);
    };

#if USE_ASYNC_LDS
    int buf = 0;
    fill_async(0, 0);
    for (int kk = 0; kk < K; kk += BK) {
        __builtin_amdgcn_s_wait_asynccnt(0);
        __syncthreads();
        if (kk + BK < K) fill_async(buf ^ 1, kk + BK);
        compute(buf);
        buf ^= 1;
    }
#else
    for (int kk = 0; kk < K; kk += BK) {
        fill_sync(0, kk);
        __syncthreads();
        compute(0);
        __syncthreads();
    }
#endif

    const int crow = (lane >> 4) * 8;
    const int ccol = lane & 15;
    #pragma unroll
    for (int mt = 0; mt < 4; ++mt) {
        #pragma unroll
        for (int nt = 0; nt < 2; ++nt) {
            const int gn = n0 + wn + nt * 16 + ccol;
            const float bv = bias[gn];
            #pragma unroll
            for (int v = 0; v < 8; ++v) {
                const int gm = m0 + wm + mt * 16 + crow + v;
                float val = acc[mt][nt][v];
                int p   = __builtin_amdgcn_cvt_pk_fp8_f32(val, val, 0, false);
                float r = __builtin_amdgcn_cvt_f32_fp8(p, 0);
                out[(size_t)gm * N + gn] = r + bv;
            }
        }
    }
}

extern "C" void kernel_launch(void* const* d_in, const int* in_sizes, int n_in,
                              void* d_out, int out_size, void* d_ws, size_t ws_size,
                              hipStream_t stream) {
    const float* x     = (const float*)d_in[0];
    const float* w     = (const float*)d_in[1];
    const float* bias  = (const float*)d_in[2];
    const float* scale = (const float*)d_in[3];
    float*       out   = (float*)d_out;

    const int K = 2048;
    const int N = 4096;
    const int M = in_sizes[0] / K;

    unsigned char* qx = (unsigned char*)d_ws;
    unsigned char* qw = qx + (size_t)M * K;

    const long long nx4 = (long long)M * K / 4;
    const long long nw4 = (long long)N * K / 4;
    fp8_quantize_kernel<<<4096, 256, 0, stream>>>(
        x, w, scale, (unsigned int*)qx, (unsigned int*)qw, nx4, nw4);

    dim3 grid(N / BN, M / BM);
    fp8_gemm_wmma_kernel<<<grid, 256, 0, stream>>>(qx, qw, bias, out, M, N, K);
}
